// GraphUNet_87797721464865
// MI455X (gfx1250) — compile-verified, matrix operand order checked by reading, not run on hardware
//
#include <hip/hip_runtime.h>
#include <hip/hip_bf16.h>
#include <stdint.h>

// Problem constants (from reference)
#define NN   50000
#define EE   400000
#define CIN  32
#define HH   64
#define KPOOL 25000          // ceil(0.5 * NN)
#define SEQL 5
#define EB   (EE + NN)       // edges of (A + I), weight 1 each (no coalescing needed)
#define PAD2 65536           // bitonic sort size (pow2 >= NN)
#define TB   256
#define WPACK (16 * 2 * 32 * 16)   // packed fragment elements per 256x64 weight matrix

typedef _Float16 half_t;
typedef __attribute__((ext_vector_type(16))) _Float16 v16h;
typedef __attribute__((ext_vector_type(8)))  float    v8f;

__device__ inline v8f wmma_f16f32(v16h a, v16h b, v8f c) {
  // D = A(16x32 f16) * B(32x16 f16) + C(16x16 f32)
  return __builtin_amdgcn_wmma_f32_16x16x32_f16(false, a, false, b, (short)0, c, false, false);
}

__device__ inline v8f vzero8() {
  v8f r;
#pragma unroll
  for (int e = 0; e < 8; e++) r[e] = 0.f;
  return r;
}

__device__ inline float sigmoidf_(float x) { return 1.f / (1.f + __expf(-x)); }

// ---------------------------------------------------------------------------
// Generic WMMA GEMM: out[rows x 64] = A[rows x K] @ W[K x 64], K in {32, 64}
// One wave -> 16 rows x 64 cols (4 col tiles). 8 waves per block (128 rows).
// ---------------------------------------------------------------------------
template <int K>
__global__ __launch_bounds__(256) void k_gemm(const float* __restrict__ A,
                                              const float* __restrict__ W,
                                              float* __restrict__ out, int rows) {
  const int lane = threadIdx.x & 31;
  const int wave = threadIdx.x >> 5;
  const int hi16 = lane >> 4;       // which 16-lane half
  const int nloc = lane & 15;
  const int rowBase = blockIdx.x * 128 + wave * 16;
  if (rowBase >= rows) return;      // wave-uniform

  const int r = rowBase + nloc;     // A-matrix row carried by this lane
  const bool valid = (r < rows);
  const float* arow = A + (size_t)(valid ? r : 0) * K;

  v16h a0, a1;
#pragma unroll
  for (int v = 0; v < 8; v++) {
    const int kb = ((v < 4) ? 0 : 16) + 8 * hi16 + 2 * (v & 3);
    a0[2 * v]     = (half_t)(valid ? arow[kb]     : 0.f);
    a0[2 * v + 1] = (half_t)(valid ? arow[kb + 1] : 0.f);
    if (K == 64) {
      a1[2 * v]     = (half_t)(valid ? arow[kb + 32] : 0.f);
      a1[2 * v + 1] = (half_t)(valid ? arow[kb + 33] : 0.f);
    }
  }

#pragma unroll
  for (int tn = 0; tn < 4; tn++) {
    const int col = tn * 16 + nloc;
    v16h b0, b1;
#pragma unroll
    for (int v = 0; v < 8; v++) {
      const int kb = ((v < 4) ? 0 : 16) + 8 * hi16 + 2 * (v & 3);
      b0[2 * v]     = (half_t)W[(size_t)kb * HH + col];
      b0[2 * v + 1] = (half_t)W[(size_t)(kb + 1) * HH + col];
      if (K == 64) {
        b1[2 * v]     = (half_t)W[(size_t)(kb + 32) * HH + col];
        b1[2 * v + 1] = (half_t)W[(size_t)(kb + 33) * HH + col];
      }
    }
    v8f c = vzero8();
    c = wmma_f16f32(a0, b0, c);
    if (K == 64) c = wmma_f16f32(a1, b1, c);
#pragma unroll
    for (int e = 0; e < 8; e++) {
      const int m = e + 8 * hi16;
      const int ro = rowBase + m;
      if (ro < rows) out[(size_t)ro * HH + col] = c[e];
    }
  }
}

// ---------------------------------------------------------------------------
// Repack a [256 x 64] f32 LSTM weight (gates-major, torch W^T convention) into
// WMMA B-fragment order: [gate_tile gt 0..15][khalf 0..1][lane 0..31][16 f16],
// 32 contiguous bytes per lane -> single aligned vector load in the LSTM.
// B[k][n] = W[gt*16 + n][k]
// ---------------------------------------------------------------------------
__global__ void k_pack_w(const float* __restrict__ W, half_t* __restrict__ out) {
  int i = blockIdx.x * TB + threadIdx.x;
  if (i >= WPACK) return;
  const int e    = i & 15;
  const int lane = (i >> 4) & 31;
  const int kh   = (i >> 9) & 1;
  const int gt   = (i >> 10) & 15;
  const int v = e >> 1, lo = e & 1;
  const int hi16 = lane >> 4, nloc = lane & 15;
  const int kb = ((v < 4) ? 0 : 16) + 8 * hi16 + 2 * (v & 3) + 32 * kh + lo;
  out[i] = (half_t)W[(size_t)(gt * 16 + nloc) * HH + kb];
}

// ---------------------------------------------------------------------------
// WMMA LSTM over sliding windows (torch gate order i,f,g,o).
// Block = 4 waves = 128 threads -> one group of 16 sequences.
// Wave w owns gate w: its 16 weight fragments (4 tiles x 2 K-halves x 2 mats)
// live in registers for the whole recurrence -- no spills, no scratch.
// Gate pre-activations, cell and hidden state go through LDS.
// scatter==0: out[row*64+c] (compact).  scatter==1: out[mi[row]*64+c].
// ---------------------------------------------------------------------------
__global__ __launch_bounds__(128) void k_lstm(const float* __restrict__ X,
                                              const int* __restrict__ mi,
                                              const int* __restrict__ Mptr,
                                              const half_t* __restrict__ WihP,
                                              const half_t* __restrict__ WhhP,
                                              const float* __restrict__ bih,
                                              const float* __restrict__ bhh,
                                              float* __restrict__ out, int scatter) {
  __shared__ float sG[4][16][HH];   // gate pre-activations (i,f,g,o)
  __shared__ float sH[16][HH];      // hidden state
  __shared__ float sC[16][HH];      // cell state
  __shared__ float sBias[4 * HH];

  const int tid  = threadIdx.x;
  const int lane = tid & 31;
  const int wave = tid >> 5;        // gate index 0..3
  const int hi16 = lane >> 4;
  const int nloc = lane & 15;
  const int M = *Mptr;
  const int rowBase = blockIdx.x * 16;
  const int rowA = rowBase + nloc;
  const int nodeA = (rowA < M) ? mi[rowA] : -1;

  for (int g = tid; g < 4 * HH; g += 128) sBias[g] = bih[g] + bhh[g];
  for (int q = tid; q < 16 * HH; q += 128) {
    sH[q >> 6][q & 63] = 0.f;
    sC[q >> 6][q & 63] = 0.f;
  }
  __syncthreads();

  // This wave's weight fragments, register-resident across all timesteps.
  v16h bi[4][2], bh[4][2];
#pragma unroll
  for (int j = 0; j < 4; j++)
#pragma unroll
    for (int kh = 0; kh < 2; kh++) {
      const int gt = wave * 4 + j;
      const size_t off = ((size_t)(gt * 2 + kh) * 32 + lane) * 16;
      bi[j][kh] = *(const v16h*)(WihP + off);
      bh[j][kh] = *(const v16h*)(WhhP + off);
    }

#pragma unroll 1
  for (int t = 0; t < SEQL; t++) {
    // ---- A fragments: x_t (global, zero-padded window) and h_{t-1} (LDS) ----
    v16h ax0, ax1, ah0, ah1;
    {
      const int tg = nodeA - (SEQL - 1) + t;
      const bool ok = (nodeA >= 0) && (tg >= 0);
      const float* xr = X + (size_t)(ok ? tg : 0) * HH;
#pragma unroll
      for (int v = 0; v < 8; v++) {
        const int kb = ((v < 4) ? 0 : 16) + 8 * hi16 + 2 * (v & 3);
        ax0[2 * v]     = (half_t)(ok ? xr[kb]      : 0.f);
        ax0[2 * v + 1] = (half_t)(ok ? xr[kb + 1]  : 0.f);
        ax1[2 * v]     = (half_t)(ok ? xr[kb + 32] : 0.f);
        ax1[2 * v + 1] = (half_t)(ok ? xr[kb + 33] : 0.f);
        ah0[2 * v]     = (half_t)sH[nloc][kb];
        ah0[2 * v + 1] = (half_t)sH[nloc][kb + 1];
        ah1[2 * v]     = (half_t)sH[nloc][kb + 32];
        ah1[2 * v + 1] = (half_t)sH[nloc][kb + 33];
      }
    }

    // ---- this wave's gate tiles: acc = bias + x@Wih^T + h@Whh^T ----
#pragma unroll
    for (int j = 0; j < 4; j++) {
      const float bv = sBias[wave * 64 + j * 16 + nloc];
      v8f acc;
#pragma unroll
      for (int e = 0; e < 8; e++) acc[e] = bv;
      acc = wmma_f16f32(ax0, bi[j][0], acc);
      acc = wmma_f16f32(ax1, bi[j][1], acc);
      acc = wmma_f16f32(ah0, bh[j][0], acc);
      acc = wmma_f16f32(ah1, bh[j][1], acc);
#pragma unroll
      for (int e = 0; e < 8; e++) sG[wave][e + 8 * hi16][j * 16 + nloc] = acc[e];
    }
    __syncthreads();

    // ---- elementwise LSTM update over the 16x64 tile ----
    for (int q = tid; q < 16 * HH; q += 128) {
      const int m = q >> 6, col = q & 63;
      const float I = sigmoidf_(sG[0][m][col]);
      const float F = sigmoidf_(sG[1][m][col]);
      const float G = tanhf(sG[2][m][col]);
      const float O = sigmoidf_(sG[3][m][col]);
      const float cn = F * sC[m][col] + I * G;
      sC[m][col] = cn;
      sH[m][col] = O * tanhf(cn);
    }
    __syncthreads();
  }

  for (int q = tid; q < 16 * HH; q += 128) {
    const int m = q >> 6, col = q & 63;
    const int row = rowBase + m;
    if (row < M) {
      const float hv = sH[m][col];
      if (scatter) out[(size_t)mi[row] * HH + col] = hv;
      else         out[(size_t)row * HH + col] = hv;
    }
  }
}

// ---------------------------------------------------------------------------
// Small helper kernels
// ---------------------------------------------------------------------------
__global__ void k_fill_f(float* p, float v, int n) {
  int i = blockIdx.x * TB + threadIdx.x; if (i < n) p[i] = v;
}
__global__ void k_fill_i(int* p, int v, int n) {
  int i = blockIdx.x * TB + threadIdx.x; if (i < n) p[i] = v;
}
__global__ void k_copy_i(int* dst, const int* src, int n) {
  int i = blockIdx.x * TB + threadIdx.x; if (i < n) dst[i] = src[i];
}
__global__ void k_rsqrt_(float* p, int n) {
  int i = blockIdx.x * TB + threadIdx.x;
  if (i < n) { float d = p[i]; p[i] = (d > 0.f) ? rsqrtf(d) : 0.f; }
}
__global__ void k_deg_edges(float* deg, const int* col, int e) {
  int i = blockIdx.x * TB + threadIdx.x; if (i < e) atomicAdd(&deg[col[i]], 1.f);
}
// out[i][f] = 2 * dinv[i]^2 * h[i][f]   (self-loop, weight 2.0 for improved GCN)
__global__ void k_self_init(float* out, const float* h, const float* dinv, int rows) {
  int i = blockIdx.x * TB + threadIdx.x;
  if (i < rows * HH) { float di = dinv[i >> 6]; out[i] = 2.f * di * di * h[i]; }
}
// message pass over explicit edge list (weight 1): out[c] += dinv[r]*dinv[c]*h[r]
__global__ void k_agg_edges(float* out, const float* __restrict__ h,
                            const float* __restrict__ dinv,
                            const int* __restrict__ row, const int* __restrict__ col, int e) {
  int idx = blockIdx.x * TB + threadIdx.x;
  int ed = idx >> 4, q = (idx & 15) * 4;
  if (ed >= e) return;
  int r = row[ed], c = col[ed];
  float s = dinv[r] * dinv[c];
  const float4 hv = *(const float4*)(h + (size_t)r * HH + q);
  float* o = out + (size_t)c * HH + q;
  atomicAdd(o + 0, s * hv.x); atomicAdd(o + 1, s * hv.y);
  atomicAdd(o + 2, s * hv.z); atomicAdd(o + 3, s * hv.w);
}
__global__ void k_bias_act(float* y, const float* b, int rows, int relu) {
  int i = blockIdx.x * TB + threadIdx.x;
  if (i < rows * HH) { float v = y[i] + b[i & 63]; y[i] = relu ? fmaxf(v, 0.f) : v; }
}
__global__ void k_wnorm(const float* w, float* outInv) {
  __shared__ float s[64];
  s[threadIdx.x] = w[threadIdx.x] * w[threadIdx.x];
  __syncthreads();
  if (threadIdx.x == 0) { float t = 0; for (int i = 0; i < 64; i++) t += s[i]; *outInv = rsqrtf(t); }
}
__global__ void k_score(const float* x1, const float* w, const float* inv, float* score, int n) {
  int i = blockIdx.x * TB + threadIdx.x;
  if (i >= n) return;
  const float* r = x1 + (size_t)i * HH;
  float s = 0.f;
  for (int f = 0; f < HH; f++) s += r[f] * w[f];
  score[i] = tanhf(s * (*inv));
}
__global__ void k_make_keys(const float* score, unsigned long long* key, int n, int total) {
  int i = blockIdx.x * TB + threadIdx.x;
  if (i >= total) return;
  if (i < n) {
    unsigned u = __float_as_uint(score[i]);
    unsigned v = (u & 0x80000000u) ? (u ^ 0xFFFFFFFFu) : (u ^ 0x80000000u); // ascending-sortable
    unsigned d = ~v;                                                         // descending score
    key[i] = ((unsigned long long)d << 32) | (unsigned)i;                    // stable tie-break
  } else key[i] = 0xFFFFFFFFFFFFFFFFull;
}
__global__ void k_bitonic(unsigned long long* key, int j, int k) {
  int i = blockIdx.x * TB + threadIdx.x;
  int l = i ^ j;
  if (l <= i) return;
  unsigned long long a = key[i], b = key[l];
  bool up = ((i & k) == 0);
  if (up ? (a > b) : (a < b)) { key[i] = b; key[l] = a; }
}
__global__ void k_perm_remap(const unsigned long long* key, int* perm, int* remap, int kkeep) {
  int i = blockIdx.x * TB + threadIdx.x;
  if (i >= kkeep) return;
  int node = (int)(key[i] & 0xFFFFFFFFull);
  perm[i] = node;
  remap[node] = i;
}
__global__ void k_gather_xp(float* xp, const float* x1, const float* score, const int* perm) {
  int i = blockIdx.x * TB + threadIdx.x;
  if (i >= KPOOL * HH) return;
  int r = i >> 6, f = i & 63;
  int node = perm[r];
  xp[i] = x1[(size_t)node * HH + f] * score[node];
}
__global__ void k_cnt_rows(int* cnt, const int* row, int e) {
  int i = blockIdx.x * TB + threadIdx.x; if (i < e) atomicAdd(&cnt[row[i]], 1);
}
// single-block chunked exclusive scan: off[0]=0, off[i+1]=sum cnt[0..i]
__global__ void k_scan_excl(const int* cnt, int* off, int n) {
  __shared__ int buf[1024];
  __shared__ int carry;
  if (threadIdx.x == 0) carry = 0;
  __syncthreads();
  for (int base = 0; base < n; base += 1024) {
    int i = base + (int)threadIdx.x;
    int v = (i < n) ? cnt[i] : 0;
    buf[threadIdx.x] = v;
    __syncthreads();
    for (int d = 1; d < 1024; d <<= 1) {
      int t = (threadIdx.x >= (unsigned)d) ? buf[threadIdx.x - d] : 0;
      __syncthreads();
      buf[threadIdx.x] += t;
      __syncthreads();
    }
    if (i < n) off[i + 1] = carry + buf[threadIdx.x];
    __syncthreads();
    if (threadIdx.x == 1023) carry += buf[1023];
    __syncthreads();
  }
  if (threadIdx.x == 0) off[0] = 0;
}
__global__ void k_fill_csr_edges(int* cpos, int* crow, int* ccol,
                                 const int* row, const int* col, int e) {
  int i = blockIdx.x * TB + threadIdx.x;
  if (i >= e) return;
  int r = row[i];
  int p = atomicAdd(&cpos[r], 1);
  crow[p] = r; ccol[p] = col[i];
}
__global__ void k_fill_csr_self(int* cpos, int* crow, int* ccol, int n) {
  int i = blockIdx.x * TB + threadIdx.x;
  if (i >= n) return;
  int p = atomicAdd(&cpos[i], 1);
  crow[p] = i; ccol[p] = i;
}
// two-hop (A+I)^2 \ diag, filtered by pooling remap: degree accumulation
__global__ void k_deg_paths(float* deg2, const int* __restrict__ remap,
                            const int* __restrict__ off, const int* __restrict__ crow,
                            const int* __restrict__ ccol, int eb) {
  int e = blockIdx.x * TB + threadIdx.x;
  if (e >= eb) return;
  int i = crow[e];
  if (remap[i] < 0) return;
  int k = ccol[e];
  int s = off[k], t = off[k + 1];
  for (int q = s; q < t; q++) {
    int j = ccol[q];
    if (j == i) continue;
    int rj = remap[j];
    if (rj >= 0) atomicAdd(&deg2[rj], 1.f);
  }
}
// two-hop message pass: x2[remap[j]] += dinv2[ri]*dinv2[rj] * hp[ri]
__global__ void k_agg_paths(float* x2, const float* __restrict__ hp,
                            const float* __restrict__ dinv2, const int* __restrict__ remap,
                            const int* __restrict__ off, const int* __restrict__ crow,
                            const int* __restrict__ ccol, int eb) {
  int e = blockIdx.x * TB + threadIdx.x;
  if (e >= eb) return;
  int i = crow[e];
  int ri = remap[i];
  if (ri < 0) return;
  int k = ccol[e];
  float di = dinv2[ri];
  const float* hr = hp + (size_t)ri * HH;
  int s = off[k], tend = off[k + 1];
  for (int q = s; q < tend; q++) {
    int j = ccol[q];
    if (j == i) continue;
    int rj = remap[j];
    if (rj < 0) continue;
    float sc = di * dinv2[rj];
    float* o = x2 + (size_t)rj * HH;
    for (int f = 0; f < HH; f++) atomicAdd(&o[f], sc * hr[f]);
  }
}
// ordered compaction of mask -> mi[], M (single block)
__global__ void k_mask_compact(const unsigned char* mask, int* mi, int* Mout, int n) {
  __shared__ int buf[1024];
  __shared__ int carry;
  if (threadIdx.x == 0) carry = 0;
  __syncthreads();
  for (int base = 0; base < n; base += 1024) {
    int i = base + (int)threadIdx.x;
    int f = (i < n && mask[i] != 0) ? 1 : 0;
    buf[threadIdx.x] = f;
    __syncthreads();
    for (int d = 1; d < 1024; d <<= 1) {
      int t = (threadIdx.x >= (unsigned)d) ? buf[threadIdx.x - d] : 0;
      __syncthreads();
      buf[threadIdx.x] += t;
      __syncthreads();
    }
    if (f) mi[carry + buf[threadIdx.x] - 1] = i;
    __syncthreads();
    if (threadIdx.x == 1023) carry += buf[1023];
    __syncthreads();
  }
  if (threadIdx.x == 0) *Mout = carry;
}
__global__ void k_scatter_h0(float* resup, const float* h0, const int* mi, const int* Mptr) {
  int i = blockIdx.x * TB + threadIdx.x;
  int r = i >> 6;
  if (r >= *Mptr) return;
  resup[(size_t)mi[r] * HH + (i & 63)] = h0[i];
}
__global__ void k_add_up(float* resup, const float* x2, const int* perm) {
  int i = blockIdx.x * TB + threadIdx.x;
  if (i >= KPOOL * HH) return;
  resup[(size_t)perm[i >> 6] * HH + (i & 63)] += x2[i];
}

// ---------------------------------------------------------------------------
extern "C" void kernel_launch(void* const* d_in, const int* in_sizes, int n_in,
                              void* d_out, int out_size, void* d_ws, size_t ws_size,
                              hipStream_t stream) {
  const float* x       = (const float*)d_in[0];
  const int*   eidx    = (const int*)d_in[1];
  const unsigned char* mask = (const unsigned char*)d_in[2];
  const float* conv0_w = (const float*)d_in[3];
  const float* conv0_b = (const float*)d_in[4];
  const float* conv1_w = (const float*)d_in[5];
  const float* conv1_b = (const float*)d_in[6];
  const float* pool_w  = (const float*)d_in[7];
  const float* up_w    = (const float*)d_in[8];
  const float* up_b    = (const float*)d_in[9];
  const float* l0_wih  = (const float*)d_in[10];
  const float* l0_whh  = (const float*)d_in[11];
  const float* l0_bih  = (const float*)d_in[12];
  const float* l0_bhh  = (const float*)d_in[13];
  const float* f_wih   = (const float*)d_in[14];
  const float* f_whh   = (const float*)d_in[15];
  const float* f_bih   = (const float*)d_in[16];
  const float* f_bhh   = (const float*)d_in[17];
  const int* erow = eidx;
  const int* ecol = eidx + EE;
  float* outp = (float*)d_out;

  // workspace carve-out (~60 MB total)
  char* p = (char*)d_ws;
  auto alloc = [&](size_t bytes) -> void* {
    void* q = (void*)p; p += (bytes + 255) & ~(size_t)255; return q;
  };
  float* x1    = (float*)alloc((size_t)NN * HH * 4);   // conv0 output (skip features / xo later)
  float* hbuf  = (float*)alloc((size_t)NN * HH * 4);   // pre-aggregation h = A@W  (reused as h0)
  float* resup = (float*)alloc((size_t)NN * HH * 4);   // res + up
  float* x2    = (float*)alloc((size_t)KPOOL * HH * 4);
  float* xp    = (float*)alloc((size_t)KPOOL * HH * 4);
  float* h0    = hbuf;                                  // lifetime-disjoint reuse
  float* dinv0 = (float*)alloc((size_t)NN * 4);
  float* dinv2 = (float*)alloc((size_t)KPOOL * 4);
  float* score = (float*)alloc((size_t)NN * 4);
  unsigned long long* keys = (unsigned long long*)alloc((size_t)PAD2 * 8);
  int* perm  = (int*)alloc((size_t)KPOOL * 4);
  int* remap = (int*)alloc((size_t)NN * 4);
  int* mi    = (int*)alloc((size_t)NN * 4);
  int* Mcnt  = (int*)alloc(256);
  int* cnt   = (int*)alloc((size_t)NN * 4);
  int* coff  = (int*)alloc((size_t)(NN + 1) * 4);
  int* cpos  = (int*)alloc((size_t)NN * 4);
  int* crow  = (int*)alloc((size_t)EB * 4);
  int* ccol  = (int*)alloc((size_t)EB * 4);
  float* winv = (float*)alloc(256);
  half_t* wp_l0i = (half_t*)alloc((size_t)WPACK * 2);  // fragment-packed LSTM weights
  half_t* wp_l0h = (half_t*)alloc((size_t)WPACK * 2);
  half_t* wp_fi  = (half_t*)alloc((size_t)WPACK * 2);
  half_t* wp_fh  = (half_t*)alloc((size_t)WPACK * 2);

  const int gN    = (NN + TB - 1) / TB;
  const int gE    = (EE + TB - 1) / TB;
  const int gNH   = (NN * HH + TB - 1) / TB;
  const int gKH   = (KPOOL * HH + TB - 1) / TB;
  const int gE16  = (EE * 16 + TB - 1) / TB;
  const int gEB   = (EB + TB - 1) / TB;
  const int gKP   = (KPOOL + TB - 1) / TB;
  const int gPad  = PAD2 / TB;
  const int gGN   = (NN + 127) / 128;     // WMMA gemm grids (16 rows/wave, 8 waves)
  const int gGK   = (KPOOL + 127) / 128;
  const int gLstm = (NN + 15) / 16;       // 16 sequences per 128-thread block
  const int gPk   = (WPACK + TB - 1) / TB;

  // ---- pre-pack LSTM weights into WMMA fragment order (f16) ----
  k_pack_w<<<gPk, TB, 0, stream>>>(l0_wih, wp_l0i);
  k_pack_w<<<gPk, TB, 0, stream>>>(l0_whh, wp_l0h);
  k_pack_w<<<gPk, TB, 0, stream>>>(f_wih,  wp_fi);
  k_pack_w<<<gPk, TB, 0, stream>>>(f_whh,  wp_fh);

  // ---- conv0: x1 = relu(GCN(x; W0,b0)) ----
  k_fill_f<<<gN, TB, 0, stream>>>(dinv0, 2.0f, NN);            // self-loop weight 2
  k_deg_edges<<<gE, TB, 0, stream>>>(dinv0, ecol, EE);
  k_rsqrt_<<<gN, TB, 0, stream>>>(dinv0, NN);
  k_gemm<32><<<gGN, TB, 0, stream>>>(x, conv0_w, hbuf, NN);    // WMMA
  k_self_init<<<gNH, TB, 0, stream>>>(x1, hbuf, dinv0, NN);
  k_agg_edges<<<gE16, TB, 0, stream>>>(x1, hbuf, dinv0, erow, ecol, EE);
  k_bias_act<<<gNH, TB, 0, stream>>>(x1, conv0_b, NN, 1);

  // ---- TopK pooling: score, stable-descending bitonic argsort, remap ----
  k_wnorm<<<1, 64, 0, stream>>>(pool_w, winv);
  k_score<<<gN, TB, 0, stream>>>(x1, pool_w, winv, score, NN);
  k_make_keys<<<gPad, TB, 0, stream>>>(score, keys, NN, PAD2);
  for (int k = 2; k <= PAD2; k <<= 1)
    for (int j = k >> 1; j > 0; j >>= 1)
      k_bitonic<<<gPad, TB, 0, stream>>>(keys, j, k);
  hipMemsetAsync(remap, 0xFF, (size_t)NN * 4, stream);         // -1
  k_perm_remap<<<gKP, TB, 0, stream>>>(keys, perm, remap, KPOOL);
  k_gather_xp<<<gKH, TB, 0, stream>>>(xp, x1, score, perm);

  // ---- CSR of B = A + I (weight 1; duplicates are aggregation-equivalent) ----
  k_fill_i<<<gN, TB, 0, stream>>>(cnt, 1, NN);                 // self loops
  k_cnt_rows<<<gE, TB, 0, stream>>>(cnt, erow, EE);
  k_scan_excl<<<1, 1024, 0, stream>>>(cnt, coff, NN);
  k_copy_i<<<gN, TB, 0, stream>>>(cpos, coff, NN);
  k_fill_csr_edges<<<gE, TB, 0, stream>>>(cpos, crow, ccol, erow, ecol, EE);
  k_fill_csr_self<<<gN, TB, 0, stream>>>(cpos, crow, ccol, NN);

  // ---- conv1 on pooled two-hop graph: x2 = relu(GCN(xp; W1,b1)) ----
  k_fill_f<<<gKP, TB, 0, stream>>>(dinv2, 2.0f, KPOOL);
  k_deg_paths<<<gEB, TB, 0, stream>>>(dinv2, remap, coff, crow, ccol, EB);
  k_rsqrt_<<<gKP, TB, 0, stream>>>(dinv2, KPOOL);
  k_gemm<64><<<gGK, TB, 0, stream>>>(xp, conv1_w, hbuf, KPOOL);  // WMMA
  k_self_init<<<gKH, TB, 0, stream>>>(x2, hbuf, dinv2, KPOOL);
  k_agg_paths<<<gEB, TB, 0, stream>>>(x2, hbuf, dinv2, remap, coff, crow, ccol, EB);
  k_bias_act<<<gKH, TB, 0, stream>>>(x2, conv1_b, KPOOL, 1);

  // ---- masked nodes + LSTM0 over sliding windows of x1 ----
  k_mask_compact<<<1, 1024, 0, stream>>>(mask, mi, Mcnt, NN);
  k_lstm<<<gLstm, 128, 0, stream>>>(x1, mi, Mcnt, wp_l0i, wp_l0h, l0_bih, l0_bhh, h0, 0); // WMMA

  // ---- res + up ----
  hipMemcpyAsync(resup, x1, (size_t)NN * HH * 4, hipMemcpyDeviceToDevice, stream);
  k_scatter_h0<<<gNH, TB, 0, stream>>>(resup, h0, mi, Mcnt);
  k_add_up<<<gKH, TB, 0, stream>>>(resup, x2, perm);

  // ---- up GCN (no activation): xo (reuse x1) ----
  k_gemm<64><<<gGN, TB, 0, stream>>>(resup, up_w, hbuf, NN);     // WMMA
  k_self_init<<<gNH, TB, 0, stream>>>(x1, hbuf, dinv0, NN);
  k_agg_edges<<<gE16, TB, 0, stream>>>(x1, hbuf, dinv0, erow, ecol, EE);
  k_bias_act<<<gNH, TB, 0, stream>>>(x1, up_b, NN, 0);

  // ---- final LSTM, scatter into output ----
  hipMemcpyAsync(outp, x1, (size_t)NN * HH * 4, hipMemcpyDeviceToDevice, stream);
  k_lstm<<<gLstm, 128, 0, stream>>>(x1, mi, Mcnt, wp_fi, wp_fh, f_bih, f_bhh, outp, 1);   // WMMA

  (void)in_sizes; (void)n_in; (void)out_size; (void)ws_size;
}
